// Attention_41223096107540
// MI455X (gfx1250) — matrix operand order repaired and checked by reading, not compiled, not run on hardware
//
#include <hip/hip_runtime.h>

typedef __attribute__((ext_vector_type(16))) _Float16 v16h;
typedef __attribute__((ext_vector_type(8)))  _Float16 v8h;
typedef __attribute__((ext_vector_type(2)))  _Float16 v2h;
typedef __attribute__((ext_vector_type(8)))  float    v8f;

#define ATTN_B 4
#define ATTN_H 16
#define ATTN_S 2048
#define ATTN_D 64
#define WAVES_PER_BLOCK 8    // 256 threads, wave32
#define KPAD 8               // f16 elements of row padding (16B) -> conflict-free b128 reads
#define VPAD 8

static __device__ __forceinline__ v16h cat8h(const _Float16* p0, const _Float16* p1) {
    v8h lo = *(const v8h*)p0;
    v8h hi = *(const v8h*)p1;
    return __builtin_shufflevector(lo, hi, 0,1,2,3,4,5,6,7,8,9,10,11,12,13,14,15);
}

static __device__ __forceinline__ v8f wmma_f16(v16h a, v16h b, v8f c) {
    return __builtin_amdgcn_wmma_f32_16x16x32_f16(
        /*neg_a=*/false, a, /*neg_b=*/false, b,
        /*c_mod=*/(short)0, c, /*reuse_a=*/false, /*reuse_b=*/false);
}

// Raw hardware exp2 (single v_exp_f32, no libm range fixup). Safe here: arguments are
// score - rowmax <= 0, and underflow-to-zero is exactly the right softmax behavior.
static __device__ __forceinline__ float fast_exp2(float x) {
    return __builtin_amdgcn_exp2f(x);
}

__global__ __launch_bounds__(WAVES_PER_BLOCK * 32)
void attn_fwd_wmma(const float* __restrict__ Q,
                   const float* __restrict__ K,
                   const float* __restrict__ V,
                   float* __restrict__ O) {
    // Double-buffered staged blocks: K as f16 row-major, V as f16 transposed (d-major).
    __shared__ _Float16 sK [2][32][ATTN_D + KPAD];
    __shared__ _Float16 sVt[2][ATTN_D][32 + VPAD];

    const int tid    = threadIdx.x;
    const int lane   = tid & 31;
    const int wave   = tid >> 5;
    const int lane15 = lane & 15;
    const int half   = lane >> 4;          // 0: lanes 0-15, 1: lanes 16-31

    const long bh   = blockIdx.y;          // fused (b,h)
    const long base = bh * (long)ATTN_S * ATTN_D;
    const float* Qb = Q + base;
    const float* Kb = K + base;
    const float* Vb = V + base;
    float*       Ob = O + base;

    const int qbase = (blockIdx.x * WAVES_PER_BLOCK + wave) * 16;

    // ---- Q as two B-fragments of Q^T (32d x 16q); B layout: elem e = K(d) = 8*half+e (e<8), 16+8*half+(e-8) (e>=8).
    // Fold softmax scale (1/sqrt(64)) AND log2(e) into Q so scores are base-2 exponents:
    // exp(s/8) == exp2(s * 0.125 * log2(e)); softmax is invariant to the base change.
    const float qscale = 0.125f * 1.44269504088896340736f;
    const float* qrow = Qb + (long)(qbase + lane15) * ATTN_D;
    v16h qf0, qf1;
#pragma unroll
    for (int i = 0; i < 16; ++i) {
        qf0[i] = (_Float16)(qrow[8 * half + i + (i >= 8 ? 8 : 0)] * qscale);
        qf1[i] = (_Float16)(qrow[32 + 8 * half + i + (i >= 8 ? 8 : 0)] * qscale);
    }

    // ---- O accumulator: four 16q x 16d C-tiles (lane = d col, VGPR = q row)
    v8f acc[4];
#pragma unroll
    for (int t = 0; t < 4; ++t) acc[t] = (v8f){0.f,0.f,0.f,0.f,0.f,0.f,0.f,0.f};

    float m_run = -3.0e38f;   // running row max, log2 domain
    float l_run = 0.0f;       // running row sum of exp2(s - m_run)

    // staging coordinates
    const int kr = tid >> 3;               // K row 0..31
    const int kc = (tid & 7) * 8;          // K col base
    const int rp = tid >> 4;               // V row pair 0..15
    const int cg = tid & 15;               // V col group 0..15

    const int d1 = 8 * half;               // A-layout first-run offset

    for (int kb = 0; kb < ATTN_S; kb += 32) {
        const int buf = (kb >> 5) & 1;

        // ======== stage K (f16 row-major) and V (f16 transposed) ========
        {
            const float* src = Kb + (long)(kb + kr) * ATTN_D + kc;
            v8h h;
#pragma unroll
            for (int i = 0; i < 8; ++i) h[i] = (_Float16)src[i];
            *(v8h*)&sK[buf][kr][kc] = h;
        }
        {
            const float* s0 = Vb + (long)(kb + 2 * rp) * ATTN_D + 4 * cg;
            const float* s1 = s0 + ATTN_D;
#pragma unroll
            for (int i = 0; i < 4; ++i) {
                v2h p; p[0] = (_Float16)s0[i]; p[1] = (_Float16)s1[i];
                *(v2h*)&sVt[buf][4 * cg + i][2 * rp] = p;
            }
        }
        if (kb + 32 < ATTN_S) {            // light prefetch of next block
            __builtin_prefetch(Kb + (long)(kb + 32 + kr) * ATTN_D + kc, 0, 3);
            __builtin_prefetch(Vb + (long)(kb + 32 + 2 * rp) * ATTN_D + 4 * cg, 0, 3);
        }
        __syncthreads();   // single barrier/iter; double buffer covers WAR across iters

        // ======== GEMM1: S^T tiles (16k x 16q), lane = q column ========
        const _Float16* k0 = &sK[buf][lane15][0];
        const _Float16* k1 = &sK[buf][16 + lane15][0];
        const v16h a00 = cat8h(k0 + d1,      k0 + 16 + d1);
        const v16h a01 = cat8h(k0 + 32 + d1, k0 + 48 + d1);
        const v16h a10 = cat8h(k1 + d1,      k1 + 16 + d1);
        const v16h a11 = cat8h(k1 + 32 + d1, k1 + 48 + d1);

        v8f st0 = (v8f){0.f,0.f,0.f,0.f,0.f,0.f,0.f,0.f};
        st0 = wmma_f16(a00, qf0, st0);
        st0 = wmma_f16(a01, qf1, st0);
        v8f st1 = (v8f){0.f,0.f,0.f,0.f,0.f,0.f,0.f,0.f};
        st1 = wmma_f16(a10, qf0, st1);
        st1 = wmma_f16(a11, qf1, st1);

        // ======== online softmax (base 2) over 32 keys for q = lane15 ========
        float mloc = -3.0e38f;
#pragma unroll
        for (int j = 0; j < 8; ++j) mloc = fmaxf(mloc, fmaxf(st0[j], st1[j]));
        mloc = fmaxf(mloc, __shfl_xor(mloc, 16, 32));

        // Rescale only when some row's max actually grew (alpha != 1). Wave-uniform branch.
        if (__ballot(mloc > m_run)) {
            const float m_new = fmaxf(m_run, mloc);
            const float alpha = fast_exp2(m_run - m_new);
            m_run = m_new;
            l_run *= alpha;
            float alphar[8];
#pragma unroll
            for (int r = 0; r < 8; ++r) alphar[r] = __shfl(alpha, r + 8 * half, 32);
#pragma unroll
            for (int t = 0; t < 4; ++t)
#pragma unroll
                for (int r = 0; r < 8; ++r) acc[t][r] *= alphar[r];
        }

        float p0[8], p1[8], rs = 0.f;
#pragma unroll
        for (int j = 0; j < 8; ++j) {
            p0[j] = fast_exp2(st0[j] - m_run);
            p1[j] = fast_exp2(st1[j] - m_run);
            rs += p0[j] + p1[j];
        }
        rs += __shfl_xor(rs, 16, 32);
        l_run += rs;

        // P^T (C layout) == P (A layout): pack directly, zero shuffles
        v16h pa;
#pragma unroll
        for (int j = 0; j < 8; ++j) { pa[j] = (_Float16)p0[j]; pa[8 + j] = (_Float16)p1[j]; }

        // ======== GEMM2: O(16q x 16d) += P(16x32) * V(32x16d) ========
#pragma unroll
        for (int t = 0; t < 4; ++t) {
            const _Float16* vp = &sVt[buf][16 * t + lane15][8 * half];
            const v16h vb = cat8h(vp, vp + 16);
            acc[t] = wmma_f16(pa, vb, acc[t]);
        }
    }

    // ======== epilogue: normalize and store (row q = r+8*half, col = 16t+lane15) ========
#pragma unroll
    for (int r = 0; r < 8; ++r) {
        const float linv = __builtin_amdgcn_rcpf(__shfl(l_run, r + 8 * half, 32));
        float* orow = Ob + (long)(qbase + r + 8 * half) * ATTN_D + lane15;
#pragma unroll
        for (int t = 0; t < 4; ++t) orow[16 * t] = acc[t][r] * linv;
    }
}

extern "C" void kernel_launch(void* const* d_in, const int* in_sizes, int n_in,
                              void* d_out, int out_size, void* d_ws, size_t ws_size,
                              hipStream_t stream) {
    const float* Q = (const float*)d_in[0];
    const float* K = (const float*)d_in[1];
    const float* V = (const float*)d_in[2];
    float*       O = (float*)d_out;
    (void)in_sizes; (void)n_in; (void)out_size; (void)d_ws; (void)ws_size;

    dim3 grid(ATTN_S / (16 * WAVES_PER_BLOCK), ATTN_B * ATTN_H);   // (16, 64)
    dim3 block(WAVES_PER_BLOCK * 32);                              // 256 threads
    attn_fwd_wmma<<<grid, block, 0, stream>>>(Q, K, V, O);
}
